// SRNN_44693429682404
// MI455X (gfx1250) — matrix operand order repaired and checked by reading, not compiled, not run on hardware
//
#include <hip/hip_runtime.h>
#include <hip/hip_bf16.h>
#include <math.h>

// ---------------------------------------------------------------------------
// Elman RNN for MI455X (gfx1250, wave32, WMMA).
//   Phase 0: convert Wx (f32 [H][I]) -> bf16 in d_ws.
//   Phase 1: xw[b,s,:] = x[b,s,:] @ Wx^T + b   via v_wmma_f32_16x16x32_bf16,
//            written straight into d_out's outputs region.
//   Phase 2: persistent scan, 2 workgroups x 16 batch rows. Wh resident in
//            VGPRs as B-fragments, h double-buffered in LDS (bf16), one
//            workgroup barrier per timestep. Overwrites xw in d_out with h.
// ---------------------------------------------------------------------------

typedef __attribute__((ext_vector_type(16))) __bf16 v16bf;
typedef __attribute__((ext_vector_type(8)))  __bf16 v8bf;
typedef __attribute__((ext_vector_type(8)))  float  v8f;
typedef __attribute__((ext_vector_type(4)))  float  v4f;

union un16 { v16bf v; v8bf h[2]; };

#define B_  32
#define S_  2048
#define I_  256
#define H_  256
#define M_  (B_ * S_)   // 65536 rows of the input-projection GEMM

// Load 8 consecutive f32 and round to a v8bf half-fragment.
__device__ __forceinline__ v8bf load8_cvt(const float* __restrict__ p) {
    v4f a = *(const v4f*)p;
    v4f b = *(const v4f*)(p + 4);
    v8bf r;
    r[0] = (__bf16)a[0]; r[1] = (__bf16)a[1]; r[2] = (__bf16)a[2]; r[3] = (__bf16)a[3];
    r[4] = (__bf16)b[0]; r[5] = (__bf16)b[1]; r[6] = (__bf16)b[2]; r[7] = (__bf16)b[3];
    return r;
}

__device__ __forceinline__ float fast_tanh(float x) {
#if __has_builtin(__builtin_amdgcn_tanhf)
    return __builtin_amdgcn_tanhf(x);
#elif __has_builtin(__builtin_amdgcn_tanh_f32)
    return __builtin_amdgcn_tanh_f32(x);
#else
    return tanhf(x);
#endif
}

__device__ __forceinline__ v8f wmma_bf16(v16bf a, v16bf b, v8f c) {
    return __builtin_amdgcn_wmma_f32_16x16x32_bf16(
        /*neg_a=*/false, a, /*neg_b=*/false, b,
        /*c_mod=*/(short)0, c, /*reuse_a=*/false, /*reuse_b=*/false);
}

// --------------------------- Phase 0: weight convert -----------------------
__global__ void srnn_cvt_wx(const float* __restrict__ w,
                            __bf16* __restrict__ o, int n) {
    int i = blockIdx.x * 256 + threadIdx.x;
    if (i < n) o[i] = (__bf16)w[i];
}

// --------------------------- Phase 1: input projection ---------------------
// Grid: 512 blocks x 256 threads. Each wave owns a 16-row M strip, full N=256.
__global__ __launch_bounds__(256) void srnn_xproj(
    const float*  __restrict__ x,      // [M_][I_] f32 (B,S flattened)
    const __bf16* __restrict__ wxbf,   // [H_][I_] bf16 row-major
    const float*  __restrict__ bias,   // [H_]
    float*        __restrict__ out)    // [M_][H_] f32 (= d_out outputs region)
{
    const int wave  = threadIdx.x >> 5;
    const int lane  = threadIdx.x & 31;
    const int lrow  = lane & 15;       // matrix row / col within half-wave
    const int hi    = lane >> 4;       // which half of the wave
    const int mbase = (blockIdx.x * 8 + wave) * 16;
    if (mbase >= M_) return;

    // A fragments for all 8 K-steps: row m = mbase+lrow of x, cvt f32->bf16.
    // Layout: lanes 0-15 hold K = ks*32+[0..15]; lanes 16-31
    // hold K = ks*32+[16..31]  (16 contiguous elems, matching B).
    v16bf afrag[8];
    const float* xrow = x + (size_t)(mbase + lrow) * I_;
#pragma unroll
    for (int ks = 0; ks < 8; ++ks) {
        const int kc0 = ks * 32 + hi * 16;
        un16 u;
        u.h[0] = load8_cvt(xrow + kc0);
        u.h[1] = load8_cvt(xrow + kc0 + 8);
        afrag[ks] = u.v;
    }

    // Sweep 16 N-tiles; B fragment = 16 contiguous bf16 from row n of Wx.
#pragma unroll 1
    for (int nt = 0; nt < 16; ++nt) {
        const int n = nt * 16 + lrow;
        const __bf16* wrow = wxbf + (size_t)n * I_;
        v8f acc = {};
#pragma unroll
        for (int ks = 0; ks < 8; ++ks) {
            const int kb = ks * 32 + hi * 16;
            un16 b;
            b.h[0] = *(const v8bf*)(wrow + kb);
            b.h[1] = *(const v8bf*)(wrow + kb + 8);
            acc = wmma_bf16(afrag[ks], b.v, acc);
        }
        const float bv = bias[n];
#pragma unroll
        for (int i = 0; i < 8; ++i) {
            const int m = mbase + i + hi * 8;  // D: VGPR i -> M = i (+8 for hi)
            out[(size_t)m * H_ + n] = acc[i] + bv;
        }
    }
}

// --------------------------- Phase 2: recurrent scan -----------------------
// Grid: 2 blocks x 256 threads. Block handles batch rows [16*blk, 16*blk+16).
// Wh B-fragments live in VGPRs (2 n-tiles x 8 k-steps x 8 VGPRs = 128/lane).
// h is exchanged through double-buffered LDS (bf16), 1 barrier per step.
__global__ __launch_bounds__(256, 1) void srnn_scan(
    const float* __restrict__ wh,      // [H_][H_] f32 row-major
    float*       __restrict__ out,     // [B_][S_][H_]: in = xw, overwritten = h
    float*       __restrict__ hlast)   // [B_][H_]
{
    __shared__ __align__(32) __bf16 hbuf[2][16 * H_];

    const int wave  = threadIdx.x >> 5;
    const int lane  = threadIdx.x & 31;
    const int lrow  = lane & 15;
    const int hi    = lane >> 4;
    const int bbase = blockIdx.x * 16;       // first batch row of this block
    const int n0    = wave * 32;             // this wave's 32 output columns

    // Resident Wh B-fragments. B[k][n] = Wh[n][k] (h @ Wh^T), so fragment =
    // 16 contiguous K from row n, K-half selected by hi.
    v16bf bfrag[2][8];
#pragma unroll
    for (int t = 0; t < 2; ++t) {
        const int n = n0 + t * 16 + lrow;
        const float* wrow = wh + (size_t)n * H_;
#pragma unroll
        for (int ks = 0; ks < 8; ++ks) {
            const int kb = ks * 32 + hi * 16;
            un16 u;
            u.h[0] = load8_cvt(wrow + kb);
            u.h[1] = load8_cvt(wrow + kb + 8);
            bfrag[t][ks] = u.v;
        }
    }

    // h0 = 0
    for (int i = threadIdx.x; i < 16 * H_; i += 256)
        hbuf[0][i] = (__bf16)0.0f;
    __syncthreads();

    for (int s = 0; s < S_; ++s) {
        const __bf16* __restrict__ hb = hbuf[s & 1];
        __bf16*       __restrict__ hw = hbuf[(s + 1) & 1];

        // Stream this step's xw early so global latency overlaps the WMMAs,
        // and prefetch next step's lines into L2.
        float xw[2][8];
#pragma unroll
        for (int t = 0; t < 2; ++t) {
            const int n = n0 + t * 16 + lrow;
#pragma unroll
            for (int i = 0; i < 8; ++i) {
                const size_t m = (size_t)(bbase + i + hi * 8);
                const float* p = out + (m * S_ + s) * H_ + n;
                xw[t][i] = *p;
                if (s + 1 < S_) __builtin_prefetch(p + H_, 0, 3);
            }
        }

        // A fragments of h from LDS (16B ds_load per chunk, 16 contiguous K per lane half, matching B).
        // Recurrent GEMM: acc[t] = h @ Wh^T for this wave's two N-tiles.
        v8f acc0 = {}, acc1 = {};
        const __bf16* arow = hb + lrow * H_;
#pragma unroll
        for (int ks = 0; ks < 8; ++ks) {
            const int kc0 = ks * 32 + hi * 16;
            un16 u;
            u.h[0] = *(const v8bf*)(arow + kc0);
            u.h[1] = *(const v8bf*)(arow + kc0 + 8);
            acc0 = wmma_bf16(u.v, bfrag[0][ks], acc0);
            acc1 = wmma_bf16(u.v, bfrag[1][ks], acc1);
        }

        // h_new = tanh(xw + acc); write f32 to out (in place over xw),
        // bf16 to the other LDS buffer for step s+1.
#pragma unroll
        for (int t = 0; t < 2; ++t) {
            const v8f acc = t ? acc1 : acc0;
            const int n = n0 + t * 16 + lrow;
#pragma unroll
            for (int i = 0; i < 8; ++i) {
                const int   r = i + hi * 8;           // local batch row 0..15
                const float v = fast_tanh(acc[i] + xw[t][i]);
                const size_t m = (size_t)(bbase + r);
                out[(m * S_ + s) * H_ + n] = v;
                hw[r * H_ + n] = (__bf16)v;
                if (s == S_ - 1) hlast[m * H_ + n] = v;
            }
        }
        __syncthreads();
    }
}

// ---------------------------------------------------------------------------
extern "C" void kernel_launch(void* const* d_in, const int* in_sizes, int n_in,
                              void* d_out, int out_size, void* d_ws, size_t ws_size,
                              hipStream_t stream) {
    const float* x    = (const float*)d_in[0];  // [B,S,I]
    const float* wx_w = (const float*)d_in[1];  // [H,I]
    const float* wx_b = (const float*)d_in[2];  // [H]
    const float* wh_w = (const float*)d_in[3];  // [H,H]

    float* out   = (float*)d_out;                       // outputs [B,S,H]
    float* hlast = out + (size_t)B_ * S_ * H_;          // then h_last [B,H]

    __bf16* wxbf = (__bf16*)d_ws;                       // 128 KB scratch

    // Phase 0: Wx f32 -> bf16 (65536 elements).
    srnn_cvt_wx<<<(H_ * I_ + 255) / 256, 256, 0, stream>>>(wx_w, wxbf, H_ * I_);

    // Phase 1: xw = x @ Wx^T + b, into d_out. 512 blocks x 8 waves x 16 rows.
    srnn_xproj<<<M_ / 128, 256, 0, stream>>>(x, wxbf, wx_b, out);

    // Phase 2: sequential scan, 2 independent half-batches.
    srnn_scan<<<2, 256, 0, stream>>>(wh_w, out, hlast);
}
